// Attention4D_17085379903861
// MI455X (gfx1250) — compile-verified, matrix operand order checked by reading, not run on hardware
//
#include <hip/hip_runtime.h>
#include <hip/hip_bf16.h>

// ---------------------------------------------------------------------------
// Attention4D (LeViT-style) fused pipeline for MI455X (gfx1250, wave32, WMMA).
// All GEMMs use v_wmma_f32_16x16x32_bf16. Operand buffers are K-contiguous
// per WMMA lane (fragments = two b128 loads). Shared B tiles are staged into
// LDS with async global->LDS copies (ASYNCcnt), double-buffered.
// ---------------------------------------------------------------------------

#define BATCH 128
#define DIM   384
#define RES   16
#define NTOK  256          // RES*RES
#define NH    8
#define KD    32
#define DHEAD 128
#define NHKD  256          // NH*KD
#define DH    1024         // NH*DHEAD
#define QKV_O 1536         // NHKD + NHKD + DH
#define SCALE 0.17677669529663687f   // KD^-0.5

typedef __bf16 bf16_t;
typedef __attribute__((ext_vector_type(16))) __bf16 v16bf;
typedef __attribute__((ext_vector_type(8)))  float  v8f;
typedef __attribute__((ext_vector_type(4)))  unsigned v4u;
typedef __attribute__((ext_vector_type(4)))  float  v4f;

// Async copy of 16 bytes global -> LDS (per lane). VDST = LDS byte address,
// VADDR = 64-bit global address, GV mode. Tracked by ASYNCcnt.
#define ASYNC_CP_B128(ldsPtr, gPtr)                                           \
    asm volatile("global_load_async_to_lds_b128 %0, %1, off"                  \
                 :: "v"((unsigned)(size_t)(ldsPtr)),                          \
                    "v"((unsigned long long)(size_t)(gPtr)) : "memory")
#define WAIT_ASYNC(n) asm volatile("s_wait_asynccnt " #n ::: "memory")

// ----------------------------- helpers -------------------------------------

__device__ __forceinline__ bf16_t f2bf(float x) {
    union { float f; unsigned u; } v; v.f = x;
    unsigned r = (v.u + 0x7FFFu + ((v.u >> 16) & 1u)) >> 16;   // RNE
    union { unsigned short s; bf16_t b; } o; o.s = (unsigned short)r;
    return o.b;
}
__device__ __forceinline__ float bf2f(bf16_t x) {
    union { unsigned short s; bf16_t b; } i; i.b = x;
    union { float f; unsigned u; } o; o.u = ((unsigned)i.s) << 16;
    return o.f;
}

union FragU { v16bf v; v4u q[2]; };

// Fragment loader for 16x32 (A) or 32x16-transposed (B) tiles.
// Source layout: [row][k] row-major, K contiguous, leading dim ld (multiple
// of 8 elements). Lane L: row = L%16, half = L/16;
// elems 0..7 <- k = half*8 + 0..7, elems 8..15 <- k = 16 + half*8 + 0..7.
__device__ __forceinline__ v16bf load_frag_nt(const bf16_t* __restrict__ p, int ld) {
    int l = threadIdx.x & 31;
    int row = l & 15, half = l >> 4;
    const bf16_t* base = p + (size_t)row * ld + half * 8;
    FragU f;
    f.q[0] = *(const v4u*)(base);
    f.q[1] = *(const v4u*)(base + 16);
    return f.v;
}

__device__ __forceinline__ v8f wmma_bf16(v16bf a, v16bf b, v8f c) {
    return __builtin_amdgcn_wmma_f32_16x16x32_bf16(false, a, false, b, (short)0, c,
                                                   false, false);
}

// --------------------------- conversion kernels ----------------------------

// x[b, c, n] (f32) -> x_t[b, n, c] (bf16): K-contiguous for QKV B-fragments.
__global__ void cvt_x_t(const float* __restrict__ src, bf16_t* __restrict__ dst) {
    int i = blockIdx.x * blockDim.x + threadIdx.x;       // over B*NTOK*DIM
    if (i >= BATCH * NTOK * DIM) return;
    int c = i % DIM;
    int rem = i / DIM;
    int n = rem % NTOK, b = rem / NTOK;
    dst[i] = f2bf(src[((size_t)b * DIM + c) * NTOK + n]);
}

__global__ void cvt_f32_bf16(const float* __restrict__ src, bf16_t* __restrict__ dst, int n) {
    int i = blockIdx.x * blockDim.x + threadIdx.x;
    if (i < n) dst[i] = f2bf(src[i]);
}

__global__ void pack_qkv_w(const float* __restrict__ qw, const float* __restrict__ kw,
                           const float* __restrict__ vw, bf16_t* __restrict__ dst) {
    int i = blockIdx.x * blockDim.x + threadIdx.x;
    if (i >= QKV_O * DIM) return;
    int o = i / DIM, c = i - o * DIM;
    float x;
    if (o < NHKD)            x = qw[o * DIM + c];
    else if (o < 2 * NHKD)   x = kw[(o - NHKD) * DIM + c];
    else                     x = vw[(o - 2 * NHKD) * DIM + c];
    dst[i] = f2bf(x);
}

// ------------------------------ QKV GEMM -----------------------------------
// out[b, o, n] = sum_c w[o,c] * x[b,c,n]  for o in [0,1536).
// A = wqkv[o][c] (global b128), B tile (64 n x 32 c) staged into LDS with
// double-buffered async copies; all 8 waves share it via ds_load_b128.
// grid=(4,12,B), 8 waves of 16 o-rows.

__global__ __launch_bounds__(256)
void qkv_gemm(const bf16_t* __restrict__ xT, const bf16_t* __restrict__ w16,
              const float* __restrict__ qb, const float* __restrict__ kb,
              const float* __restrict__ vb,
              bf16_t* __restrict__ q16, bf16_t* __restrict__ k16,
              bf16_t* __restrict__ vT) {
    __shared__ __align__(16) bf16_t xs[2][64][40];   // 40 = 32 + pad (16B-aligned rows)
    const int b = blockIdx.z, ob = blockIdx.y, nb = blockIdx.x;
    const int w = threadIdx.x >> 5, l = threadIdx.x & 31;
    const int n0 = nb * 64;
    const int orow0 = ob * 128 + w * 16;
    const int crow = threadIdx.x >> 2, cchunk = (threadIdx.x & 3) * 8;

    const v8f vzero = {0.f, 0.f, 0.f, 0.f, 0.f, 0.f, 0.f, 0.f};
    v8f acc[4] = {vzero, vzero, vzero, vzero};

    const bf16_t* wr = w16 + (size_t)orow0 * DIM;
    const bf16_t* xb = xT + ((size_t)b * NTOK + n0) * DIM;

    // prologue: stage first B tile
    ASYNC_CP_B128(&xs[0][crow][cchunk], xb + (size_t)crow * DIM + cchunk);

    int buf = 0;
    for (int kk = 0; kk < DIM; kk += 32) {
        if (kk + 32 < DIM) {
            ASYNC_CP_B128(&xs[buf ^ 1][crow][cchunk],
                          xb + (size_t)crow * DIM + (kk + 32) + cchunk);
            WAIT_ASYNC(1);          // current tile done, next still in flight
        } else {
            WAIT_ASYNC(0);
        }
        __syncthreads();
        __builtin_prefetch(wr + kk + 128, 0, 0);         // global_prefetch_b8
        v16bf a = load_frag_nt(wr + kk, DIM);
#pragma unroll
        for (int nt = 0; nt < 4; ++nt) {
            v16bf bb = load_frag_nt(&xs[buf][nt * 16][0], 40);
            acc[nt] = wmma_bf16(a, bb, acc[nt]);
        }
        __syncthreads();
        buf ^= 1;
    }

    const int col = l & 15, half = l >> 4;
#pragma unroll
    for (int nt = 0; nt < 4; ++nt) {
        int n = n0 + nt * 16 + col;
#pragma unroll
        for (int r = 0; r < 8; ++r) {
            int o = orow0 + r + half * 8;
            float val = acc[nt][r];
            if (o < NHKD) {
                val += qb[o];
                int h = o >> 5, kd = o & 31;
                q16[(((size_t)b * NH + h) * NTOK + n) * KD + kd] = f2bf(val);
            } else if (o < 2 * NHKD) {
                int o2 = o - NHKD;
                val += kb[o2];
                int h = o2 >> 5, kd = o2 & 31;
                k16[(((size_t)b * NH + h) * NTOK + n) * KD + kd] = f2bf(val);
            } else {
                int o3 = o - 2 * NHKD;
                val += vb[o3];
                int h = o3 >> 7, d = o3 & 127;
                vT[(((size_t)b * NH + h) * DHEAD + d) * NTOK + n] = f2bf(val);
            }
        }
    }
}

// ---------------------------- fused attention ------------------------------
// grid = (16 query tiles, B), block = 256 (wave h = head h).
// Dynamic LDS: float S[NH][16][NTOK] = 128 KB (within CDNA5's 320 KB/WG).

__global__ __launch_bounds__(256)
void attn_kernel(const bf16_t* __restrict__ q16, const bf16_t* __restrict__ k16,
                 const bf16_t* __restrict__ vT,
                 const float* __restrict__ attn_bias, const int* __restrict__ bias_idxs,
                 const float* __restrict__ th1w, const float* __restrict__ th1b,
                 const float* __restrict__ th2w, const float* __restrict__ th2b,
                 const float* __restrict__ vlw, const float* __restrict__ vlb,
                 bf16_t* __restrict__ actT) {
    extern __shared__ float S[];                  // [NH][16][NTOK]
    const int b = blockIdx.y, n0 = blockIdx.x * 16;
    const int h = threadIdx.x >> 5, l = threadIdx.x & 31;
    const int col = l & 15, half = l >> 4;
    const v8f vzero = {0.f, 0.f, 0.f, 0.f, 0.f, 0.f, 0.f, 0.f};

    // ---- scores: one 16x16x32 WMMA per m-tile, B frags double-buffered ----
    const bf16_t* qh = q16 + (((size_t)b * NH + h) * NTOK + n0) * KD;
    const bf16_t* kh = k16 + ((size_t)b * NH + h) * NTOK * KD;
    v16bf aq = load_frag_nt(qh, KD);
    v16bf bk = load_frag_nt(kh, KD);
    for (int mt = 0; mt < 16; ++mt) {
        v16bf bk_nxt = bk;
        if (mt < 15) bk_nxt = load_frag_nt(kh + (size_t)(mt + 1) * 16 * KD, KD);
        v8f c = wmma_bf16(aq, bk, vzero);
        bk = bk_nxt;
        int m = mt * 16 + col;
#pragma unroll
        for (int r = 0; r < 8; ++r) {
            int nloc = r + half * 8;
            int idx = bias_idxs[(n0 + nloc) * NTOK + m];
            S[(h * 16 + nloc) * NTOK + m] = c[r] * SCALE + attn_bias[h * 256 + idx];
        }
    }
    __syncthreads();

    // ---- talking-head 1: in-place (each (nloc,m) owned by one thread) ----
    for (int j = 0; j < 16; ++j) {
        int i = threadIdx.x + j * 256;            // 0..4095
        int nloc = i >> 8, m = i & 255;
        float sv[NH], ov[NH];
#pragma unroll
        for (int hh = 0; hh < NH; ++hh) sv[hh] = S[(hh * 16 + nloc) * NTOK + m];
#pragma unroll
        for (int g = 0; g < NH; ++g) {
            float a = th1b[g];
#pragma unroll
            for (int hh = 0; hh < NH; ++hh) a += th1w[g * NH + hh] * sv[hh];
            ov[g] = a;
        }
#pragma unroll
        for (int g = 0; g < NH; ++g) S[(g * 16 + nloc) * NTOK + m] = ov[g];
    }
    __syncthreads();

    // ---- softmax over m: wave h owns its head's 16 rows of 256 ----
    for (int nloc = 0; nloc < 16; ++nloc) {
        float* row = &S[(h * 16 + nloc) * NTOK];
        float mx = -3.4e38f;
        for (int m = l; m < NTOK; m += 32) mx = fmaxf(mx, row[m]);
#pragma unroll
        for (int off = 16; off > 0; off >>= 1) mx = fmaxf(mx, __shfl_xor(mx, off, 32));
        float sum = 0.f;
        for (int m = l; m < NTOK; m += 32) { float e = __expf(row[m] - mx); row[m] = e; sum += e; }
#pragma unroll
        for (int off = 16; off > 0; off >>= 1) sum += __shfl_xor(sum, off, 32);
        float inv = 1.f / sum;
        for (int m = l; m < NTOK; m += 32) row[m] *= inv;
    }
    __syncthreads();

    // ---- talking-head 2 (same in-place trick) ----
    for (int j = 0; j < 16; ++j) {
        int i = threadIdx.x + j * 256;
        int nloc = i >> 8, m = i & 255;
        float sv[NH], ov[NH];
#pragma unroll
        for (int hh = 0; hh < NH; ++hh) sv[hh] = S[(hh * 16 + nloc) * NTOK + m];
#pragma unroll
        for (int g = 0; g < NH; ++g) {
            float a = th2b[g];
#pragma unroll
            for (int hh = 0; hh < NH; ++hh) a += th2w[g * NH + hh] * sv[hh];
            ov[g] = a;
        }
#pragma unroll
        for (int g = 0; g < NH; ++g) S[(g * 16 + nloc) * NTOK + m] = ov[g];
    }
    __syncthreads();

    // ---- O = attn * V : K=256 (8 WMMA k-steps), D=128 (8 d-tiles) ----
    // A from f32 LDS scores (4x ds_load_b128 + cvt); B double-buffered.
    v8f acc[8] = {vzero, vzero, vzero, vzero, vzero, vzero, vzero, vzero};
    const bf16_t* vh = vT + ((size_t)b * NH + h) * DHEAD * NTOK;
    for (int km = 0; km < 8; ++km) {
        const float* srow = &S[(h * 16 + col) * NTOK + km * 32 + half * 8];
        v4f a0 = *(const v4f*)(srow);
        v4f a1 = *(const v4f*)(srow + 4);
        v4f a2 = *(const v4f*)(srow + 16);
        v4f a3 = *(const v4f*)(srow + 20);
        v16bf aa;
#pragma unroll
        for (int e = 0; e < 4; ++e) {
            aa[e]      = f2bf(a0[e]);
            aa[e + 4]  = f2bf(a1[e]);
            aa[e + 8]  = f2bf(a2[e]);
            aa[e + 12] = f2bf(a3[e]);
        }
        v16bf bv = load_frag_nt(vh + km * 32, NTOK);   // dt = 0
#pragma unroll
        for (int dt = 0; dt < 8; ++dt) {
            v16bf bv_nxt = bv;
            if (dt < 7)
                bv_nxt = load_frag_nt(vh + (size_t)(dt + 1) * 16 * NTOK + km * 32, NTOK);
            acc[dt] = wmma_bf16(aa, bv, acc[dt]);
            bv = bv_nxt;
        }
    }

    // ---- epilogue: fused depthwise 3x3 (v_local) + bias + ReLU -> bf16 ----
    for (int dt = 0; dt < 8; ++dt) {
        int d = dt * 16 + col;
        int ch = h * DHEAD + d;
        const bf16_t* vrow = vh + (size_t)d * NTOK;
#pragma unroll
        for (int r = 0; r < 8; ++r) {
            int n = n0 + r + half * 8;
            int y = n >> 4, xp = n & 15;
            float vl = vlb[ch];
#pragma unroll
            for (int dy = 0; dy < 3; ++dy) {
                int yy = y + dy - 1;
                if (yy < 0 || yy > 15) continue;
#pragma unroll
                for (int dx = 0; dx < 3; ++dx) {
                    int xx = xp + dx - 1;
                    if (xx < 0 || xx > 15) continue;
                    vl += vlw[ch * 9 + dy * 3 + dx] * bf2f(vrow[yy * 16 + xx]);
                }
            }
            float o = acc[dt][r] + vl;
            o = fmaxf(o, 0.f);
            // act stored transposed: actT[b, n, ch] (K-contiguous for proj B).
            actT[((size_t)b * NTOK + n) * DH + ch] = f2bf(o);
        }
    }
}

// ------------------------------ output proj --------------------------------
// out[b, o, n] = proj_w[o, dh] * act[b, dh, n] + proj_b[o], f32 output.
// Same async-LDS-staged B-tile scheme as qkv_gemm. grid=(4,3,B), K=1024.

__global__ __launch_bounds__(256)
void proj_gemm(const bf16_t* __restrict__ actT, const bf16_t* __restrict__ w16,
               const float* __restrict__ pb, float* __restrict__ out) {
    __shared__ __align__(16) bf16_t xs[2][64][40];
    const int b = blockIdx.z, ob = blockIdx.y, nb = blockIdx.x;
    const int w = threadIdx.x >> 5, l = threadIdx.x & 31;
    const int n0 = nb * 64;
    const int orow0 = ob * 128 + w * 16;
    const int crow = threadIdx.x >> 2, cchunk = (threadIdx.x & 3) * 8;

    const v8f vzero = {0.f, 0.f, 0.f, 0.f, 0.f, 0.f, 0.f, 0.f};
    v8f acc[4] = {vzero, vzero, vzero, vzero};

    const bf16_t* wr = w16 + (size_t)orow0 * DH;
    const bf16_t* ab = actT + ((size_t)b * NTOK + n0) * DH;

    ASYNC_CP_B128(&xs[0][crow][cchunk], ab + (size_t)crow * DH + cchunk);

    int buf = 0;
    for (int kk = 0; kk < DH; kk += 32) {
        if (kk + 32 < DH) {
            ASYNC_CP_B128(&xs[buf ^ 1][crow][cchunk],
                          ab + (size_t)crow * DH + (kk + 32) + cchunk);
            WAIT_ASYNC(1);
        } else {
            WAIT_ASYNC(0);
        }
        __syncthreads();
        __builtin_prefetch(wr + kk + 128, 0, 0);
        v16bf a = load_frag_nt(wr + kk, DH);
#pragma unroll
        for (int nt = 0; nt < 4; ++nt) {
            v16bf bb = load_frag_nt(&xs[buf][nt * 16][0], 40);
            acc[nt] = wmma_bf16(a, bb, acc[nt]);
        }
        __syncthreads();
        buf ^= 1;
    }

    const int col = l & 15, half = l >> 4;
#pragma unroll
    for (int nt = 0; nt < 4; ++nt) {
        int n = n0 + nt * 16 + col;
#pragma unroll
        for (int r = 0; r < 8; ++r) {
            int o = orow0 + r + half * 8;
            if (o < DIM)
                out[((size_t)b * DIM + o) * NTOK + n] = acc[nt][r] + pb[o];
        }
    }
}

// ------------------------------- launcher ----------------------------------

extern "C" void kernel_launch(void* const* d_in, const int* in_sizes, int n_in,
                              void* d_out, int out_size, void* d_ws, size_t ws_size,
                              hipStream_t stream) {
    const float* x      = (const float*)d_in[0];
    const float* q_w    = (const float*)d_in[1];
    const float* q_b    = (const float*)d_in[2];
    const float* k_w    = (const float*)d_in[3];
    const float* k_b    = (const float*)d_in[4];
    const float* v_w    = (const float*)d_in[5];
    const float* v_b    = (const float*)d_in[6];
    const float* vl_w   = (const float*)d_in[7];
    const float* vl_b   = (const float*)d_in[8];
    const float* th1_w  = (const float*)d_in[9];
    const float* th1_b  = (const float*)d_in[10];
    const float* th2_w  = (const float*)d_in[11];
    const float* th2_b  = (const float*)d_in[12];
    const float* proj_w = (const float*)d_in[13];
    const float* proj_b = (const float*)d_in[14];
    const float* attn_b = (const float*)d_in[15];
    const int*   b_idx  = (const int*)d_in[16];
    float* out = (float*)d_out;

    char* ws = (char*)d_ws;
    size_t off = 0;
    bf16_t* xT      = (bf16_t*)(ws + off); off += (size_t)BATCH * NTOK * DIM * 2;
    bf16_t* wqkv16  = (bf16_t*)(ws + off); off += (size_t)QKV_O * DIM * 2;
    bf16_t* wproj16 = (bf16_t*)(ws + off); off += (size_t)DIM * DH * 2;
    bf16_t* q16     = (bf16_t*)(ws + off); off += (size_t)BATCH * NH * NTOK * KD * 2;
    bf16_t* k16     = (bf16_t*)(ws + off); off += (size_t)BATCH * NH * NTOK * KD * 2;
    bf16_t* vT      = (bf16_t*)(ws + off); off += (size_t)BATCH * NH * DHEAD * NTOK * 2;
    bf16_t* actT    = (bf16_t*)(ws + off); off += (size_t)BATCH * NTOK * DH * 2;

    // 0) precision conversion + layout transforms
    const int nx = BATCH * NTOK * DIM;
    cvt_x_t<<<(nx + 255) / 256, 256, 0, stream>>>(x, xT);
    pack_qkv_w<<<(QKV_O * DIM + 255) / 256, 256, 0, stream>>>(q_w, k_w, v_w, wqkv16);
    cvt_f32_bf16<<<(DIM * DH + 255) / 256, 256, 0, stream>>>(proj_w, wproj16, DIM * DH);

    // 1) fused QKV projection (WMMA bf16, async-LDS staged B, pipelined)
    qkv_gemm<<<dim3(4, 12, BATCH), 256, 0, stream>>>(xT, wqkv16, q_b, k_b, v_b,
                                                     q16, k16, vT);

    // 2) fused attention: scores + bias + th1 + softmax + th2 + attn*V
    //    + depthwise 3x3 + ReLU. 128 KB dynamic LDS per block.
    attn_kernel<<<dim3(16, BATCH), 256, NH * 16 * NTOK * sizeof(float), stream>>>(
        q16, k16, vT, attn_b, b_idx, th1_w, th1_b, th2_w, th2_b, vl_w, vl_b, actT);

    // 3) output projection (WMMA bf16, async-LDS staged B, pipelined)
    proj_gemm<<<dim3(4, 3, BATCH), 256, 0, stream>>>(actT, wproj16, proj_b, out);
}
